// CrossAttention_56813827392247
// MI455X (gfx1250) — compile-verified, matrix operand order checked by reading, not run on hardware
//
#include <hip/hip_runtime.h>
#include <hip/hip_bf16.h>

typedef __bf16 bf16;
typedef __attribute__((ext_vector_type(16))) __bf16 v16bf;
typedef __attribute__((ext_vector_type(8)))  float  v8f;

#define CA_B 4
#define CA_S 1024
#define CA_T 576
#define CA_D 1024
#define CA_H 16
#define CA_DH 64

__device__ __forceinline__ bf16 f2bf(float f) {
  unsigned u = __builtin_bit_cast(unsigned, f);
  unsigned r = u + 0x7FFFu + ((u >> 16) & 1u);   // round-to-nearest-even
  unsigned short h = (unsigned short)(r >> 16);
  return __builtin_bit_cast(bf16, h);
}
__device__ __forceinline__ bf16 toBf(float v) { return f2bf(v); }
__device__ __forceinline__ bf16 toBf(bf16 v)  { return v; }
__device__ __forceinline__ void storeOut(float* p, float v) { *p = v; }
__device__ __forceinline__ void storeOut(bf16*  p, float v) { *p = f2bf(v); }

// ---- CDNA5 WMMA operand loaders (wave32, 16x16x32 bf16) ----
// A operand (16x32): lane holds row = lane&15; per ISA 16-bit A layout the 16
// bf16 values are K = kb + {0..7} and K = 16 + kb + {0..7}, kb = (lane>>4)*8.
__device__ __forceinline__ v16bf lds_load_A(const bf16* base, int ld, int lane) {
  const bf16* p = base + (lane & 15) * ld + ((lane >> 4) << 3);
  v16bf a;
#pragma unroll
  for (int i = 0; i < 8; ++i) { a[i] = p[i]; a[i + 8] = p[16 + i]; }
  return a;
}
// B operand (32x16), data stored [k][n] row-major with leading dim ldn;
// lane holds col n = lane&15, same K striping as A (B mirrors A per ISA).
__device__ __forceinline__ v16bf lds_load_B(const bf16* base, int ldn, int lane) {
  const int n  = lane & 15;
  const int kb = (lane >> 4) << 3;
  v16bf b;
#pragma unroll
  for (int i = 0; i < 8; ++i) {
    b[i]     = base[(kb + i) * ldn + n];
    b[i + 8] = base[(16 + kb + i) * ldn + n];
  }
  return b;
}

__device__ __forceinline__ v8f wmma_bf16(v16bf a, v16bf b, v8f c) {
  return __builtin_amdgcn_wmma_f32_16x16x32_bf16(false, a, false, b,
                                                 (short)0, c, false, false);
}

// ============================================================================
// Tiled GEMM: C[M,N] = A[M,K] @ W[:, :N] + bias  (W leading dim = ldw)
// 128 threads = 4 waves; block tile 128x64; each wave computes 32x64 (2x4
// C-tiles) so every B operand feeds two WMMAs. K staged in 32-chunks via LDS.
// ============================================================================
template <typename Tin, typename Tout>
__global__ __launch_bounds__(128) void linear_wmma(
    const Tin* __restrict__ A, const float* __restrict__ W,
    const float* __restrict__ bias, Tout* __restrict__ C,
    int M, int N, int K, int ldw) {
  __shared__ bf16 sA[128 * 32];
  __shared__ bf16 sW[32 * 64];

  const int tid  = threadIdx.x;
  const int lane = tid & 31;
  const int wave = tid >> 5;
  const int n0 = blockIdx.x * 64;
  const int m0 = blockIdx.y * 128;
  const int wr = wave * 32;              // this wave's 32-row group

  v8f acc[2][4];
#pragma unroll
  for (int mt = 0; mt < 2; ++mt)
#pragma unroll
    for (int nt = 0; nt < 4; ++nt)
#pragma unroll
      for (int r = 0; r < 8; ++r) acc[mt][nt][r] = 0.f;

  const int wk = tid >> 2, wn = (tid & 3) * 16;   // W staging: 32 rows x 64

  for (int k0 = 0; k0 < K; k0 += 32) {
    __syncthreads();
#pragma unroll
    for (int i = 0; i < 32; ++i)        // A staging: one full row per thread
      sA[tid * 32 + i] = toBf(A[(size_t)(m0 + tid) * K + k0 + i]);
#pragma unroll
    for (int i = 0; i < 16; ++i)
      sW[wk * 64 + wn + i] = f2bf(W[(size_t)(k0 + wk) * ldw + n0 + wn + i]);
    if (k0 + 32 < K) {   // gfx1250 global_prefetch of next tiles
      __builtin_prefetch(&A[(size_t)(m0 + tid) * K + k0 + 32], 0, 0);
      __builtin_prefetch(&W[(size_t)(k0 + 32 + wk) * ldw + n0 + wn], 0, 0);
    }
    __syncthreads();

    v16bf a0 = lds_load_A(sA + wr * 32, 32, lane);
    v16bf a1 = lds_load_A(sA + (wr + 16) * 32, 32, lane);
#pragma unroll
    for (int nt = 0; nt < 4; ++nt) {
      v16bf b = lds_load_B(sW + nt * 16, 64, lane);
      acc[0][nt] = wmma_bf16(a0, b, acc[0][nt]);
      acc[1][nt] = wmma_bf16(a1, b, acc[1][nt]);
    }
  }

  const int half = lane >> 4;
#pragma unroll
  for (int nt = 0; nt < 4; ++nt) {
    const int col = n0 + nt * 16 + (lane & 15);
    const float bv = bias[col];
#pragma unroll
    for (int mt = 0; mt < 2; ++mt)
#pragma unroll
      for (int r = 0; r < 8; ++r) {
        const int row = m0 + wr + mt * 16 + r + half * 8;
        storeOut(&C[(size_t)row * N + col], acc[mt][nt][r] + bv);
      }
  }
}

// ============================================================================
// Flash-style cross attention. One block = one (b,h) x 64 query rows.
// 128 threads = 4 waves, each wave owns 16 query rows.
// Q [B*S, D] bf16, KV [B*T, D] bf16 (used as both K and V), O [B*S, D] bf16.
// ============================================================================
__global__ __launch_bounds__(128) void attn_wmma(
    const bf16* __restrict__ Q, const bf16* __restrict__ KV,
    bf16* __restrict__ O) {
  __shared__ bf16 qt[64 * 64];          // Q tile   [s][d]
  __shared__ bf16 kt[64 * 64];          // KV tile  [t][d]
  __shared__ bf16 pt[4 * 16 * 64];      // per-wave softmax probs [row][t]

  const int tid  = threadIdx.x;
  const int lane = tid & 31;
  const int wave = tid >> 5;
  const int half = lane >> 4;
  const int bh = blockIdx.x;
  const int b  = bh / CA_H;
  const int h  = bh % CA_H;
  const int s0 = blockIdx.y * 64;
  const int wr = wave * 16;

  // stage Q tile once (bf16, 16B vector copies; offsets are 64B-aligned)
  {
    const int r = tid >> 1, cb = (tid & 1) * 32;
    const uint4* src = (const uint4*)(Q + (size_t)(b * CA_S + s0 + r) * CA_D + h * CA_DH + cb);
    uint4* dst = (uint4*)(qt + r * 64 + cb);
#pragma unroll
    for (int i = 0; i < 4; ++i) dst[i] = src[i];
  }
  __syncthreads();
  // Q operands are invariant over the whole T loop: load them exactly once.
  v16bf aq[2];
  aq[0] = lds_load_A(qt + wr * 64, 64, lane);
  aq[1] = lds_load_A(qt + wr * 64 + 32, 64, lane);

  v8f o[4];
  float mrow[8], lrow[8];
#pragma unroll
  for (int nt = 0; nt < 4; ++nt)
#pragma unroll
    for (int r = 0; r < 8; ++r) o[nt][r] = 0.f;
#pragma unroll
  for (int r = 0; r < 8; ++r) { mrow[r] = -3.0e38f; lrow[r] = 0.f; }

  for (int tb = 0; tb < CA_T / 64; ++tb) {
    __syncthreads();                    // previous iter done with kt
    {
      const int r = tid >> 1, cb = (tid & 1) * 32;
      const uint4* src = (const uint4*)(KV + (size_t)(b * CA_T + tb * 64 + r) * CA_D + h * CA_DH + cb);
      uint4* dst = (uint4*)(kt + r * 64 + cb);
#pragma unroll
      for (int i = 0; i < 4; ++i) dst[i] = src[i];
    }
    __syncthreads();

    // scores: S[16 x 64] = Qw (16 x dh) @ KVblk^T (dh x 64), dh = 64 = 2 K-chunks
    v8f sc[4];
#pragma unroll
    for (int nt = 0; nt < 4; ++nt) {
#pragma unroll
      for (int r = 0; r < 8; ++r) sc[nt][r] = 0.f;
#pragma unroll
      for (int kc = 0; kc < 2; ++kc) {
        // B[k=d][n=t] = kt[t][d] -> transposed access == A-pattern on kt rows
        v16bf bm = lds_load_A(kt + (nt * 16) * 64 + kc * 32, 64, lane);
        sc[nt] = wmma_bf16(aq[kc], bm, sc[nt]);
      }
#pragma unroll
      for (int r = 0; r < 8; ++r) sc[nt][r] *= 0.125f;   // 1/sqrt(64)
    }

    // online softmax; row stats live in registers (slot r <-> row wr+r+half*8)
    float alpha[8];
#pragma unroll
    for (int r = 0; r < 8; ++r) {
      float mx = fmaxf(fmaxf(sc[0][r], sc[1][r]), fmaxf(sc[2][r], sc[3][r]));
      mx = fmaxf(mx, __shfl_xor(mx, 1, 32));
      mx = fmaxf(mx, __shfl_xor(mx, 2, 32));
      mx = fmaxf(mx, __shfl_xor(mx, 4, 32));
      mx = fmaxf(mx, __shfl_xor(mx, 8, 32));
      const float mnew = fmaxf(mrow[r], mx);
      float sum = 0.f;
#pragma unroll
      for (int nt = 0; nt < 4; ++nt) {
        const float p = __expf(sc[nt][r] - mnew);
        sc[nt][r] = p;
        sum += p;
      }
      sum += __shfl_xor(sum, 1, 32);
      sum += __shfl_xor(sum, 2, 32);
      sum += __shfl_xor(sum, 4, 32);
      sum += __shfl_xor(sum, 8, 32);
      const float a_ = __expf(mrow[r] - mnew);
      lrow[r] = lrow[r] * a_ + sum;
      mrow[r] = mnew;
      alpha[r] = a_;
    }
#pragma unroll
    for (int nt = 0; nt < 4; ++nt)
#pragma unroll
      for (int r = 0; r < 8; ++r) o[nt][r] *= alpha[r];

    // stage probs (C layout -> row-major bf16) into this wave's pane
    bf16* pw = pt + wave * 16 * 64;
#pragma unroll
    for (int nt = 0; nt < 4; ++nt)
#pragma unroll
      for (int r = 0; r < 8; ++r)
        pw[(r + half * 8) * 64 + nt * 16 + (lane & 15)] = f2bf(sc[nt][r]);
    __syncthreads();

    // O += P (16 x 64) @ KVblk (64 x dh); P operands loaded once per T-block
    v16bf ap0 = lds_load_A(pw, 64, lane);
    v16bf ap1 = lds_load_A(pw + 32, 64, lane);
#pragma unroll
    for (int nt = 0; nt < 4; ++nt) {
      v16bf b0 = lds_load_B(kt + nt * 16, 64, lane);
      v16bf b1 = lds_load_B(kt + 32 * 64 + nt * 16, 64, lane);
      o[nt] = wmma_bf16(ap0, b0, o[nt]);
      o[nt] = wmma_bf16(ap1, b1, o[nt]);
    }
  }

  // normalize and write merged-head output (bf16)
#pragma unroll
  for (int nt = 0; nt < 4; ++nt) {
    const int d = nt * 16 + (lane & 15);
#pragma unroll
    for (int r = 0; r < 8; ++r) {
      const int s = s0 + wr + r + half * 8;
      O[(size_t)(b * CA_S + s) * CA_D + h * CA_DH + d] = f2bf(o[nt][r] / lrow[r]);
    }
  }
}

// ============================================================================
extern "C" void kernel_launch(void* const* d_in, const int* in_sizes, int n_in,
                              void* d_out, int out_size, void* d_ws, size_t ws_size,
                              hipStream_t stream) {
  (void)in_sizes; (void)n_in; (void)out_size; (void)ws_size;
  const float* x      = (const float*)d_in[0];   // [B,S,D]
  const float* enc    = (const float*)d_in[1];   // [B,T,D]
  const float* w_attn = (const float*)d_in[2];   // [D,3D]
  const float* b_attn = (const float*)d_in[3];   // [3D]
  const float* w_vis  = (const float*)d_in[4];   // [D,D]
  const float* b_vis  = (const float*)d_in[5];   // [D]
  const float* w_proj = (const float*)d_in[6];   // [D,D]
  const float* b_proj = (const float*)d_in[7];   // [D]
  float* out = (float*)d_out;                    // [B,S,D]

  const int MS = CA_B * CA_S;   // 4096
  const int MT = CA_B * CA_T;   // 2304 (= 18 * 128)

  bf16* Qb  = (bf16*)d_ws;                    // [MS, D]
  bf16* KVb = Qb + (size_t)MS * CA_D;         // [MT, D]
  bf16* Ab  = KVb + (size_t)MT * CA_D;        // [MS, D]

  // 1) Q = x @ w_attn[:, :D]  (q slice of qkv; k/v of c_attn are unused)
  linear_wmma<float, bf16><<<dim3(CA_D / 64, MS / 128), 128, 0, stream>>>(
      x, w_attn, b_attn, Qb, MS, CA_D, CA_D, 3 * CA_D);
  // 2) KV = enc @ w_vis + b_vis  (shared key/value projection)
  linear_wmma<float, bf16><<<dim3(CA_D / 64, MT / 128), 128, 0, stream>>>(
      enc, w_vis, b_vis, KVb, MT, CA_D, CA_D, CA_D);
  // 3) attention
  attn_wmma<<<dim3(CA_B * CA_H, CA_S / 64), 128, 0, stream>>>(Qb, KVb, Ab);
  // 4) out = A @ w_proj + b_proj
  linear_wmma<bf16, float><<<dim3(CA_D / 64, MS / 128), 128, 0, stream>>>(
      Ab, w_proj, b_proj, out, MS, CA_D, CA_D, CA_D);
}
